// MultiHeadLatentAttention_25142738551219
// MI455X (gfx1250) — matrix operand order repaired and checked by reading, not compiled, not run on hardware
//
#include <hip/hip_runtime.h>

typedef __bf16 bf16_t;
typedef __attribute__((ext_vector_type(16))) __bf16 v16bf;
typedef __attribute__((ext_vector_type(8)))  __bf16 v8bf;
typedef __attribute__((ext_vector_type(8)))  float   v8f;

constexpr int BB = 4, NN = 2048, CC = 1024, HH = 16, DD = 64, RR = 64;
constexpr int BN = BB * NN;                 // 8192 token rows
constexpr float ATTN_SCALE = 0.125f;        // 1/sqrt(64)

// ---------------- helpers ----------------

__device__ __forceinline__ bf16_t f2bf(float f) {
  unsigned u = __float_as_uint(f);
  unsigned r = (u + 0x7FFFu + ((u >> 16) & 1u)) >> 16;   // round-to-nearest-even
  unsigned short s = (unsigned short)r;
  return __builtin_bit_cast(__bf16, s);
}

__device__ __forceinline__ v8f zero8() {
  v8f z;
#pragma unroll
  for (int i = 0; i < 8; ++i) z[i] = 0.f;
  return z;
}

// A operand (16x32 bf16, M x K): lane holds row = lane&15,
// K chunks at k0 + 8*(lane>>4) + {0..7} and + {16..23}.
__device__ __forceinline__ v16bf load_a16(const bf16_t* base, int ld, int lane, int k0) {
  int row = lane & 15;
  int h   = (lane >> 4) << 3;  // 0 or 8
  const bf16_t* p = base + (size_t)row * ld + k0 + h;
  v8bf lo = *(const v8bf*)(p);
  v8bf hi = *(const v8bf*)(p + 16);
  v16bf r;
#pragma unroll
  for (int i = 0; i < 8; ++i) { r[i] = lo[i]; r[i + 8] = hi[i]; }
  return r;
}

// B operand (32x16 bf16, K x N), loaded from B^T stored row-major (N x K):
// lane holds column = lane&15, K chunks at k0 + 8*(lane>>4) + {0..7} and + {16..23}.
__device__ __forceinline__ v16bf load_b16(const bf16_t* base, int ld, int lane, int k0) {
  int col = lane & 15;
  int h   = (lane >> 4) << 3;  // 0 or 8
  const bf16_t* p = base + (size_t)col * ld + k0 + h;
  v8bf lo = *(const v8bf*)(p);
  v8bf hi = *(const v8bf*)(p + 16);
  v16bf r;
#pragma unroll
  for (int i = 0; i < 8; ++i) { r[i] = lo[i]; r[i + 8] = hi[i]; }
  return r;
}

__device__ __forceinline__ v8f wmma_bf16(v16bf a, v16bf b, v8f c) {
  return __builtin_amdgcn_wmma_f32_16x16x32_bf16(false, a, false, b, (short)0, c, false, false);
}

// ---------------- kernel 0: f32 -> bf16 convert ----------------

__global__ void cvt_f32_bf16(const float* __restrict__ s, bf16_t* __restrict__ d, int n) {
  int i = blockIdx.x * blockDim.x + threadIdx.x;
  if (i < n) d[i] = f2bf(s[i]);
}

// ---------------- kernel 1: latents + rmsnorm ----------------
// qlat[BN,64], kvlat[BN,64] = rmsnorm(x @ Wa^T).  Wave = 16-row tile, 8 col-tiles
// (cols 0..63 = q latent via w_q_a, 64..127 = kv latent via w_kv_a), K=1024.

__global__ void latent_kernel(const bf16_t* __restrict__ xb,
                              const bf16_t* __restrict__ wqa,   // [64,1024]
                              const bf16_t* __restrict__ wkva,  // [64,1024]
                              const float* __restrict__ qw,
                              const float* __restrict__ kvw,
                              bf16_t* __restrict__ qlat,
                              bf16_t* __restrict__ kvlat) {
  __shared__ float S[4][16][128];
  int lane = threadIdx.x & 31;
  int wave = threadIdx.x >> 5;
  int row0 = (blockIdx.x * 4 + wave) * 16;

  v8f acc[8];
#pragma unroll
  for (int t = 0; t < 8; ++t) acc[t] = zero8();

  const bf16_t* arow = xb + (size_t)row0 * CC;
  for (int k0 = 0; k0 < CC; k0 += 32) {
    v16bf a = load_a16(arow, CC, lane, k0);
#pragma unroll
    for (int t = 0; t < 8; ++t) {
      const bf16_t* wb = (t < 4) ? (wqa + (size_t)(t * 16) * CC)
                                 : (wkva + (size_t)((t - 4) * 16) * CC);
      v16bf b = load_b16(wb, CC, lane, k0);
      acc[t] = wmma_bf16(a, b, acc[t]);
    }
  }

  int n = lane & 15, mh = (lane >> 4) << 3;
#pragma unroll
  for (int t = 0; t < 8; ++t)
#pragma unroll
    for (int v = 0; v < 8; ++v) S[wave][mh + v][t * 16 + n] = acc[t][v];
  __syncthreads();

  // rmsnorm: lanes 0..15 -> q rows, lanes 16..31 -> kv rows
  int r = lane & 15;
  bool isq = lane < 16;
  float* srow = &S[wave][r][isq ? 0 : 64];
  float ssq = 0.f;
#pragma unroll 8
  for (int j = 0; j < 64; ++j) ssq += srow[j] * srow[j];
  float sc = rsqrtf(ssq * (1.f / 64.f) + 1e-6f);
  const float* nw = isq ? qw : kvw;
  bf16_t* dst = (isq ? qlat : kvlat) + (size_t)(row0 + r) * RR;
#pragma unroll 8
  for (int j = 0; j < 64; ++j) dst[j] = f2bf(srow[j] * sc * nw[j]);
}

// ---------------- kernel 2a: q expansion ----------------
// q[B,H,N,D] (bf16) = qlat @ w_q_b^T ; wave = one 16x16 tile, K=64.

__global__ void q_expand_kernel(const bf16_t* __restrict__ qlat,
                                const bf16_t* __restrict__ wqb,   // [1024,64]
                                bf16_t* __restrict__ qh) {
  int lane = threadIdx.x & 31, wave = threadIdx.x >> 5;
  int ct = blockIdx.x;                 // 0..63
  int rt = blockIdx.y * 4 + wave;      // 0..511
  int row0 = rt * 16, col0 = ct * 16;

  v8f acc = zero8();
#pragma unroll
  for (int ks = 0; ks < 2; ++ks) {
    v16bf a = load_a16(qlat + (size_t)row0 * RR, RR, lane, ks * 32);
    v16bf b = load_b16(wqb + (size_t)col0 * RR, RR, lane, ks * 32);
    acc = wmma_bf16(a, b, acc);
  }
  int n = lane & 15, mh = (lane >> 4) << 3;
  int col = col0 + n, h = col >> 6, d = col & 63;
#pragma unroll
  for (int v = 0; v < 8; ++v) {
    int row = row0 + mh + v;
    int b0 = row >> 11, nq = row & (NN - 1);
    qh[(((size_t)(b0 * HH + h)) * NN + nq) * DD + d] = f2bf(acc[v]);
  }
}

// ---------------- kernel 2b: kv expansion ----------------
// k[B,H,N,D] and vT[B,H,D,N] (bf16) from kvlat @ w_kv_b^T (cols: h*128 + {k:0..63, v:64..127})

__global__ void kv_expand_kernel(const bf16_t* __restrict__ kvlat,
                                 const bf16_t* __restrict__ wkvb,  // [2048,64]
                                 bf16_t* __restrict__ kh,
                                 bf16_t* __restrict__ vT) {
  int lane = threadIdx.x & 31, wave = threadIdx.x >> 5;
  int ct = blockIdx.x;                 // 0..127
  int rt = blockIdx.y * 4 + wave;      // 0..511
  int row0 = rt * 16, col0 = ct * 16;

  v8f acc = zero8();
#pragma unroll
  for (int ks = 0; ks < 2; ++ks) {
    v16bf a = load_a16(kvlat + (size_t)row0 * RR, RR, lane, ks * 32);
    v16bf b = load_b16(wkvb + (size_t)col0 * RR, RR, lane, ks * 32);
    acc = wmma_bf16(a, b, acc);
  }
  int n = lane & 15, mh = (lane >> 4) << 3;
  int col = col0 + n, h = col >> 7, rem = col & 127;
#pragma unroll
  for (int v = 0; v < 8; ++v) {
    int row = row0 + mh + v;
    int b0 = row >> 11, nq = row & (NN - 1);
    bf16_t val = f2bf(acc[v]);
    if (rem < 64)
      kh[(((size_t)(b0 * HH + h)) * NN + nq) * DD + rem] = val;
    else
      vT[(((size_t)(b0 * HH + h)) * DD + (rem - 64)) * NN + nq] = val;
  }
}

// ---------------- kernel 3: flash attention ----------------
// Wave = 16 query rows of one (b,h); online softmax; Q in regs; S/P staged in LDS.

__global__ void attn_kernel(const bf16_t* __restrict__ qh,
                            const bf16_t* __restrict__ kh,
                            const bf16_t* __restrict__ vT,
                            bf16_t* __restrict__ yb) {
  __shared__ float  Sls[4][16][64];
  __shared__ bf16_t Pls[4][16][64];
  __shared__ float  Als[4][16];
  __shared__ float  Lls[4][16];

  int lane = threadIdx.x & 31, wave = threadIdx.x >> 5;
  int b = blockIdx.z, h = blockIdx.y;
  int q0 = blockIdx.x * 64 + wave * 16;
  int n = lane & 15, mh = (lane >> 4) << 3;

  const bf16_t* qbase = qh + (((size_t)(b * HH + h)) * NN + q0) * DD;
  const bf16_t* kbase = kh + ((size_t)(b * HH + h)) * NN * DD;
  const bf16_t* vbase = vT + ((size_t)(b * HH + h)) * DD * NN;

  v16bf aq0 = load_a16(qbase, DD, lane, 0);
  v16bf aq1 = load_a16(qbase, DD, lane, 32);

  v8f accY[4];
#pragma unroll
  for (int t = 0; t < 4; ++t) accY[t] = zero8();
  float m_i = -1e30f, l_i = 0.f;   // live in lanes 0..15 (lane == row)

  for (int j0 = 0; j0 < NN; j0 += 64) {
    // ---- scores S = Q K^T for 64 keys (4 j-tiles x 2 k-steps) ----
#pragma unroll
    for (int jt = 0; jt < 4; ++jt) {
      const bf16_t* kb = kbase + (size_t)(j0 + jt * 16) * DD;
      v8f s = zero8();
      s = wmma_bf16(aq0, load_b16(kb, DD, lane, 0),  s);
      s = wmma_bf16(aq1, load_b16(kb, DD, lane, 32), s);
#pragma unroll
      for (int v = 0; v < 8; ++v) Sls[wave][mh + v][jt * 16 + n] = s[v];
    }
    __syncthreads();

    // ---- online softmax: lane r owns query row r ----
    if (lane < 16) {
      float* srow = Sls[wave][lane];
      float mx = -1e30f;
#pragma unroll 8
      for (int j = 0; j < 64; ++j) mx = fmaxf(mx, srow[j] * ATTN_SCALE);
      float mnew  = fmaxf(m_i, mx);
      float alpha = __expf(m_i - mnew);
      float sum = 0.f;
#pragma unroll 8
      for (int j = 0; j < 64; ++j) {
        float p = __expf(srow[j] * ATTN_SCALE - mnew);
        sum += p;
        Pls[wave][lane][j] = f2bf(p);
      }
      l_i = l_i * alpha + sum;
      m_i = mnew;
      Als[wave][lane] = alpha;
    }
    __syncthreads();

    // ---- rescale Y accumulators by per-row alpha ----
    float av[8];
#pragma unroll
    for (int v = 0; v < 8; ++v) av[v] = Als[wave][mh + v];
#pragma unroll
    for (int t = 0; t < 4; ++t)
#pragma unroll
      for (int v = 0; v < 8; ++v) accY[t][v] *= av[v];

    // ---- Y += P @ V  (P from LDS as A-operand, V^T rows as B-operand) ----
#pragma unroll
    for (int ks = 0; ks < 2; ++ks) {
      v16bf ap = load_a16(&Pls[wave][0][0], 64, lane, ks * 32);
#pragma unroll
      for (int t = 0; t < 4; ++t) {
        v16bf bv = load_b16(vbase + (size_t)(t * 16) * NN + j0, NN, lane, ks * 32);
        accY[t] = wmma_bf16(ap, bv, accY[t]);
      }
    }
    __syncthreads();
  }

  if (lane < 16) Lls[wave][lane] = 1.f / l_i;
  __syncthreads();
  float lv[8];
#pragma unroll
  for (int v = 0; v < 8; ++v) lv[v] = Lls[wave][mh + v];

  size_t grow0 = (size_t)b * NN + q0;
#pragma unroll
  for (int t = 0; t < 4; ++t)
#pragma unroll
    for (int v = 0; v < 8; ++v)
      yb[(grow0 + mh + v) * (HH * DD) + h * DD + t * 16 + n] = f2bf(accY[t][v] * lv[v]);
}

// ---------------- kernel 4: output projection ----------------
// out[BN,1024] fp32 = yb @ w_proj^T ; wave = 16x64 tile, K=1024.

__global__ void proj_kernel(const bf16_t* __restrict__ yb,
                            const bf16_t* __restrict__ wp,   // [1024,1024]
                            float* __restrict__ out) {
  int lane = threadIdx.x & 31, wave = threadIdx.x >> 5;
  int row0 = (blockIdx.y * 4 + wave) * 16;
  int col0 = blockIdx.x * 64;

  v8f acc[4];
#pragma unroll
  for (int t = 0; t < 4; ++t) acc[t] = zero8();

  const bf16_t* arow = yb + (size_t)row0 * CC;
  for (int k0 = 0; k0 < CC; k0 += 32) {
    v16bf a = load_a16(arow, CC, lane, k0);
#pragma unroll
    for (int t = 0; t < 4; ++t) {
      v16bf b = load_b16(wp + (size_t)(col0 + t * 16) * CC, CC, lane, k0);
      acc[t] = wmma_bf16(a, b, acc[t]);
    }
  }
  int n = lane & 15, mh = (lane >> 4) << 3;
#pragma unroll
  for (int t = 0; t < 4; ++t)
#pragma unroll
    for (int v = 0; v < 8; ++v)
      out[(size_t)(row0 + mh + v) * CC + col0 + t * 16 + n] = acc[t][v];
}

// ---------------- launcher ----------------

extern "C" void kernel_launch(void* const* d_in, const int* in_sizes, int n_in,
                              void* d_out, int out_size, void* d_ws, size_t ws_size,
                              hipStream_t stream) {
  const float* x         = (const float*)d_in[0];
  const float* w_kv_a    = (const float*)d_in[1];
  const float* w_kv_b    = (const float*)d_in[2];
  const float* w_q_a     = (const float*)d_in[3];
  const float* w_q_b     = (const float*)d_in[4];
  const float* w_proj    = (const float*)d_in[5];
  const float* kv_norm_w = (const float*)d_in[6];
  const float* q_norm_w  = (const float*)d_in[7];
  float* out = (float*)d_out;

  char* p = (char*)d_ws;
  auto alloc = [&](size_t bytes) -> void* {
    void* r = (void*)p;
    p += (bytes + 255) & ~(size_t)255;
    return r;
  };
  bf16_t* xb    = (bf16_t*)alloc((size_t)BN * CC * 2);
  bf16_t* wqab  = (bf16_t*)alloc((size_t)RR * CC * 2);
  bf16_t* wkvab = (bf16_t*)alloc((size_t)RR * CC * 2);
  bf16_t* wqbb  = (bf16_t*)alloc((size_t)CC * RR * 2);
  bf16_t* wkvbb = (bf16_t*)alloc((size_t)2 * CC * RR * 2);
  bf16_t* wpb   = (bf16_t*)alloc((size_t)CC * CC * 2);
  bf16_t* qlat  = (bf16_t*)alloc((size_t)BN * RR * 2);
  bf16_t* kvlat = (bf16_t*)alloc((size_t)BN * RR * 2);
  bf16_t* qhb   = (bf16_t*)alloc((size_t)BN * HH * DD * 2);
  bf16_t* khb   = (bf16_t*)alloc((size_t)BN * HH * DD * 2);
  bf16_t* vTb   = (bf16_t*)alloc((size_t)BN * HH * DD * 2);
  bf16_t* ybb   = (bf16_t*)alloc((size_t)BN * HH * DD * 2);

  auto cvt = [&](const float* s, bf16_t* d, int nelem) {
    cvt_f32_bf16<<<(nelem + 255) / 256, 256, 0, stream>>>(s, d, nelem);
  };
  cvt(x,      xb,    BN * CC);
  cvt(w_q_a,  wqab,  RR * CC);
  cvt(w_kv_a, wkvab, RR * CC);
  cvt(w_q_b,  wqbb,  CC * RR);
  cvt(w_kv_b, wkvbb, 2 * CC * RR);
  cvt(w_proj, wpb,   CC * CC);

  latent_kernel<<<BN / 64, 128, 0, stream>>>(xb, wqab, wkvab, q_norm_w, kv_norm_w,
                                             qlat, kvlat);
  q_expand_kernel<<<dim3(64, BN / 64), 128, 0, stream>>>(qlat, wqbb, qhb);
  kv_expand_kernel<<<dim3(128, BN / 64), 128, 0, stream>>>(kvlat, wkvbb, khb, vTb);
  attn_kernel<<<dim3(NN / 64, HH, BB), 128, 0, stream>>>(qhb, khb, vTb, ybb);
  proj_kernel<<<dim3(CC / 64, BN / 64), 128, 0, stream>>>(ybb, wpb, out);
}
